// PointNetPlusPlus_22273700397327
// MI455X (gfx1250) — matrix operand order repaired and checked by reading, not compiled, not run on hardware
//
#include <hip/hip_runtime.h>
#include <stdint.h>

// PointNet++-style pipeline for gfx1250 (MI455X), wave32 + WMMA f16->f32.
// B=32, N=2048, NSAMPLE=4, embed 42, hidden 60, conv 122->30->60->2.

#define Bsz 32
#define Npts 2048
#define ROWS1 (Bsz * Npts)          // 65536
#define ROWS2 (Bsz * Npts * 4)      // 262144
#define EPS_BN 1e-5f

typedef __attribute__((ext_vector_type(16))) _Float16 v16h;
typedef __attribute__((ext_vector_type(8)))  float    v8f;
typedef __attribute__((ext_vector_type(4)))  int      v4i;

#if defined(__has_builtin)
#if __has_builtin(__builtin_amdgcn_global_load_async_to_lds_b128)
#define HAVE_ASYNC_LDS 1
#endif
#endif

// workspace layout (bytes); total ~41 MB
static constexpr size_t OFF_H     = 0;                    // 65536*64*4 = 16777216 (padded stride 64)
static constexpr size_t OFF_HW    = 16777216;             // 65536*32*4 =  8388608 (padded stride 32)
static constexpr size_t OFF_GTERM = 25165824;             // 32*30*4
static constexpr size_t OFF_IDX   = 25169920;             // 65536*4*4  =  1048576
static constexpr size_t OFF_Z0    = 26218496;             // 262144*32*2= 16777216
static constexpr size_t OFF_W1H   = 42995712;             // 64*64 f16  = 8192
static constexpr size_t OFF_W0H   = 43003904;             // 32*64 f16  = 4096
static constexpr size_t OFF_W1CH  = 43008000;             // 64*32 f16  = 4096
static constexpr size_t OFF_PRM   = 43012096;             // 10*64 f32

// Fourier feature k of (x0,x1); k in [0,42), else 0. Turn-reduced native sin.
__device__ __forceinline__ float embed_feat(float x0, float x1, int k) {
  if (k >= 42) return 0.0f;
  if (k == 0)  return x0;
  if (k == 1)  return x1;
  int t = k - 2;
  int f = t >> 2, r = t & 3;
  float v = ((r & 1) ? x1 : x0) * (float)(1 << f);
  // sin for r<2; cos(v) = sin shifted a quarter turn
  float rev = __builtin_fmaf(v, 0.15915494309189535f, (r < 2) ? 0.0f : 0.25f);
  rev = rev - rintf(rev);
  return __sinf(rev * 6.283185307179586f);
}

// ---------------- kernel 0: pack weights/params, zero-padded --------------
__global__ void k_pack(const float* __restrict__ W1,  const float* __restrict__ b1,
                       const float* __restrict__ g1,  const float* __restrict__ be1,
                       const float* __restrict__ Wc0, const float* __restrict__ Wc1,
                       const float* __restrict__ bc1, const float* __restrict__ gc1,
                       const float* __restrict__ bec1,const float* __restrict__ g2,
                       const float* __restrict__ be2, const float* __restrict__ W4,
                       _Float16* __restrict__ W1h, _Float16* __restrict__ W0h,
                       _Float16* __restrict__ W1ch, float* __restrict__ prm) {
  const float rs = rsqrtf(1.0f + EPS_BN);
  const int tid = threadIdx.x;
  for (int i = tid; i < 64 * 64; i += 256) {
    int n = i >> 6, k = i & 63;
    W1h[i] = (_Float16)((n < 60 && k < 42) ? W1[n * 42 + k] : 0.0f);
  }
  for (int i = tid; i < 32 * 64; i += 256) {
    int n = i >> 6, k = i & 63;
    W0h[i] = (_Float16)((n < 30 && k < 60) ? Wc0[n * 122 + 2 + k] : 0.0f);
  }
  for (int i = tid; i < 64 * 32; i += 256) {
    int n = i >> 5, k = i & 31;
    W1ch[i] = (_Float16)((n < 60 && k < 30) ? Wc1[n * 30 + k] : 0.0f);
  }
  if (tid < 64) {
    int c = tid; bool v = c < 60;
    prm[0 * 64 + c] = v ? b1[c]       : 0.0f;
    prm[1 * 64 + c] = v ? g1[c] * rs  : 0.0f;
    prm[2 * 64 + c] = v ? be1[c]      : 0.0f;
    prm[3 * 64 + c] = v ? bc1[c]      : 0.0f;
    prm[4 * 64 + c] = v ? gc1[c] * rs : 0.0f;
    prm[5 * 64 + c] = v ? bec1[c]     : 0.0f;
    prm[6 * 64 + c] = v ? g2[c] * rs  : 0.0f;
    prm[7 * 64 + c] = v ? be2[c]      : 0.0f;
    prm[8 * 64 + c] = v ? W4[c]       : 0.0f;
    prm[9 * 64 + c] = v ? W4[60 + c]  : 0.0f;
  }
}

// ---------------- kernel 1: h = bn(relu(embed(x) @ W1^T + b1)) ------------
// One wave per 16-row M tile; 4 N tiles; h stored padded (stride 64),
// cols 60..63 exact zeros (padded params give 0 epilogue). Branch-free.
__global__ void k_embed_gemm1(const float* __restrict__ x,
                              const _Float16* __restrict__ W1h,
                              const float* __restrict__ prm,
                              float* __restrict__ h) {
  const int lane  = threadIdx.x & 31;
  const int mtile = blockIdx.x * 4 + (threadIdx.x >> 5);
  const int m  = lane & 15;
  const int hi = lane >> 4;
  const int row0 = mtile * 16;
  const int row  = row0 + m;
  const float2 p = ((const float2*)x)[row];

  const int kb = hi * 8;
  v16h a0, a1;
  for (int e = 0; e < 8; ++e) {
    a0[e]     = (_Float16)embed_feat(p.x, p.y,      kb + e);
    a0[e + 8] = (_Float16)embed_feat(p.x, p.y, 16 + kb + e);
    a1[e]     = (_Float16)embed_feat(p.x, p.y, 32 + kb + e);
    a1[e + 8] = (_Float16)embed_feat(p.x, p.y, 48 + kb + e);
  }

  const int kbase = hi * 8;
  for (int t = 0; t < 4; ++t) {
    const int n = t * 16 + m;
    const _Float16* wr = W1h + n * 64;
    v16h bv0 = __builtin_shufflevector(*(const _Float16 __attribute__((ext_vector_type(8)))*)(wr + kbase), *(const _Float16 __attribute__((ext_vector_type(8)))*)(wr + 16 + kbase), 0,1,2,3,4,5,6,7,8,9,10,11,12,13,14,15);
    v16h bv1 = __builtin_shufflevector(*(const _Float16 __attribute__((ext_vector_type(8)))*)(wr + 32 + kbase), *(const _Float16 __attribute__((ext_vector_type(8)))*)(wr + 48 + kbase), 0,1,2,3,4,5,6,7,8,9,10,11,12,13,14,15);
    v8f c = {};
    c = __builtin_amdgcn_wmma_f32_16x16x32_f16(false, a0, false, bv0, (short)0, c, false, false);
    c = __builtin_amdgcn_wmma_f32_16x16x32_f16(false, a1, false, bv1, (short)0, c, false, false);

    const int col = t * 16 + m;
    const float bb = prm[col];
    const float sc = prm[64 + col];
    const float of = prm[128 + col];
    for (int v = 0; v < 8; ++v) {
      int r = row0 + v + 8 * hi;
      float val = fmaxf(c[v] + bb, 0.0f);
      h[(size_t)r * 64 + col] = val * sc + of;   // exact 0 for col>=60
    }
  }
}

// ---------------- kernel 2: gmax over N; gterm = gmax @ Wc0[:,62:]^T + bc0
__global__ void k_gmax_gterm(const float* __restrict__ h,
                             const float* __restrict__ Wc0,
                             const float* __restrict__ bc0,
                             float* __restrict__ gterm) {
  __shared__ float gm[60];
  const int b = blockIdx.x;
  const int c = threadIdx.x;
  if (c < 60) {
    float mx = -3.0e38f;
    const float* hb = h + (size_t)b * Npts * 64 + c;
    for (int i = 0; i < Npts; ++i) mx = fmaxf(mx, hb[i * 64]);
    gm[c] = mx;
  }
  __syncthreads();
  if (c < 30) {
    float acc = bc0[c];
    for (int k = 0; k < 60; ++k) acc += gm[k] * Wc0[c * 122 + 62 + k];
    gterm[b * 30 + c] = acc;
  }
}

// ---------------- kernel 3: ball query (first 4 indices with d^2 <= 1) ---
__global__ void k_ballquery(const float* __restrict__ x, int* __restrict__ idx) {
  __shared__ float2 pts[Npts];  // 16 KB: whole batch point set in LDS
  const int b = blockIdx.x >> 3;
  const int chunk = blockIdx.x & 7;
  const float2* xb = (const float2*)(x + (size_t)b * Npts * 2);
#if defined(HAVE_ASYNC_LDS)
  {
    typedef __attribute__((address_space(1))) v4i g_v4i;
    typedef __attribute__((address_space(3))) v4i l_v4i;
    for (int t = threadIdx.x; t < Npts / 2; t += 256) {   // 16 B per op
      g_v4i* gp = (g_v4i*)(uintptr_t)((const char*)xb + (size_t)t * 16);
      l_v4i* lp = (l_v4i*)(unsigned)(uintptr_t)((char*)pts + (size_t)t * 16);
      __builtin_amdgcn_global_load_async_to_lds_b128(gp, lp, 0, 0);
    }
#if __has_builtin(__builtin_amdgcn_s_wait_asynccnt)
    __builtin_amdgcn_s_wait_asynccnt(0);
#else
    asm volatile("s_wait_asynccnt 0x0" ::: "memory");
#endif
  }
#else
  for (int t = threadIdx.x; t < Npts; t += 256) pts[t] = xb[t];
#endif
  __syncthreads();
  const int i = chunk * 256 + threadIdx.x;
  const float px = pts[i].x, py = pts[i].y;
  int nb0 = 0, nb1 = 0, nb2 = 0, nb3 = 0, cnt = 0;
  for (int j = 0; j < Npts; ++j) {
    float dx = pts[j].x - px, dy = pts[j].y - py;
    if (dx * dx + dy * dy <= 1.0f) {
      if (cnt == 0) nb0 = j; else if (cnt == 1) nb1 = j;
      else if (cnt == 2) nb2 = j; else nb3 = j;
      if (++cnt == 4) break;
    }
  }
  if (cnt < 2) nb1 = nb0;
  if (cnt < 3) nb2 = nb0;
  if (cnt < 4) nb3 = nb0;
  const int base = ((b * Npts) + i) * 4;
  idx[base + 0] = nb0; idx[base + 1] = nb1;
  idx[base + 2] = nb2; idx[base + 3] = nb3;
}

// ---------------- kernel 4: hW = h @ Wc0[:,2:62]^T (65536x60 -> x30) -----
// Branch-free: h cols 60..63 are zero, W0h rows 30,31 zero -> hW cols 30,31
// are exact zeros; stores unconditional at stride 32.
__global__ void k_hW(const float* __restrict__ h,
                     const _Float16* __restrict__ W0h,
                     float* __restrict__ hW) {
  const int lane  = threadIdx.x & 31;
  const int mtile = blockIdx.x * 4 + (threadIdx.x >> 5);
  const int m  = lane & 15;
  const int hi = lane >> 4;
  const int row0 = mtile * 16;
  const float* hr = h + (size_t)(row0 + m) * 64;

  const int kb = hi * 8;
  v16h a0, a1;
  for (int e = 0; e < 8; ++e) {
    a0[e]     = (_Float16)hr[kb + e];
    a0[e + 8] = (_Float16)hr[16 + kb + e];
    a1[e]     = (_Float16)hr[32 + kb + e];
    a1[e + 8] = (_Float16)hr[48 + kb + e];
  }

  const int kbase = hi * 8;
  for (int t = 0; t < 2; ++t) {
    const int n = t * 16 + m;
    const _Float16* wr = W0h + n * 64;
    v16h bv0 = __builtin_shufflevector(*(const _Float16 __attribute__((ext_vector_type(8)))*)(wr + kbase), *(const _Float16 __attribute__((ext_vector_type(8)))*)(wr + 16 + kbase), 0,1,2,3,4,5,6,7,8,9,10,11,12,13,14,15);
    v16h bv1 = __builtin_shufflevector(*(const _Float16 __attribute__((ext_vector_type(8)))*)(wr + 32 + kbase), *(const _Float16 __attribute__((ext_vector_type(8)))*)(wr + 48 + kbase), 0,1,2,3,4,5,6,7,8,9,10,11,12,13,14,15);
    v8f c = {};
    c = __builtin_amdgcn_wmma_f32_16x16x32_f16(false, a0, false, bv0, (short)0, c, false, false);
    c = __builtin_amdgcn_wmma_f32_16x16x32_f16(false, a1, false, bv1, (short)0, c, false, false);
    const int col = t * 16 + m;
    for (int v = 0; v < 8; ++v)
      hW[(size_t)(row0 + v + 8 * hi) * 32 + col] = c[v];
  }
}

// ---------------- kernel 5: z0 = relu(bn(gx.Wc0[:, :2] + hW[idx] + gterm))
// stored f16, row stride 32 (cols 30,31 zero) -> direct WMMA A operand.
__global__ void k_group_z0(const float* __restrict__ x,
                           const int* __restrict__ idx,
                           const float* __restrict__ hW,
                           const float* __restrict__ gterm,
                           const float* __restrict__ Wc0,
                           const float* __restrict__ gc0,
                           const float* __restrict__ bec0,
                           _Float16* __restrict__ z0) {
  __shared__ float sw0[30], sw1[30], ssc[30], sof[30], sgt[30];
  const int gid = blockIdx.x * 256 + threadIdx.x;  // (b,i,s) flattened
  const int pi  = gid >> 2;                        // b*N + i
  const int b   = pi >> 11;
  if (threadIdx.x < 30) {
    const int c = threadIdx.x;
    const float rs = rsqrtf(1.0f + EPS_BN);
    sw0[c] = Wc0[c * 122 + 0];
    sw1[c] = Wc0[c * 122 + 1];
    ssc[c] = gc0[c] * rs;
    sof[c] = bec0[c];
    sgt[c] = gterm[b * 30 + c];   // one block spans a single batch b
  }
  __syncthreads();

  const int j  = idx[gid];
  const int pj = (b << 11) + j;
  const float2 xi = ((const float2*)x)[pi];
  const float2 xj = ((const float2*)x)[pj];
  const float g0 = xj.x - xi.x;
  const float g1 = xj.y - xi.y;
  const float* hwr = hW + (size_t)pj * 32;
  _Float16* zr = z0 + (size_t)gid * 32;
  for (int c = 0; c < 30; ++c) {
    float pre = g0 * sw0[c] + g1 * sw1[c] + hwr[c] + sgt[c];
    zr[c] = (_Float16)fmaxf(pre * ssc[c] + sof[c], 0.0f);
  }
  zr[30] = (_Float16)0.0f;
  zr[31] = (_Float16)0.0f;
}

// ---------------- kernel 6: z1 = relu(bn(z0@Wc1^T+bc1)); max over the 4
// samples (4 consecutive accumulator VGPRs); z2 = relu(bn2); out = z2@W4^T+b4
__global__ void k_z1_out(const _Float16* __restrict__ z0,
                         const _Float16* __restrict__ W1ch,
                         const float* __restrict__ prm,
                         const float* __restrict__ b4,
                         float* __restrict__ out) {
  const int lane  = threadIdx.x & 31;
  const int mtile = blockIdx.x * 4 + (threadIdx.x >> 5);
  const int m  = lane & 15;
  const int hi = lane >> 4;
  const int row = mtile * 16 + m;  // row over B*N*4 (point,sample)

  const _Float16* zr = z0 + (size_t)row * 32;
  const int kb = hi * 8;
  v16h a;
  for (int e = 0; e < 8; ++e) { a[e] = zr[kb + e]; a[e + 8] = zr[16 + kb + e]; }

  const int kbase = hi * 8;
  float accA0 = 0.f, accA1 = 0.f, accB0 = 0.f, accB1 = 0.f;
  for (int t = 0; t < 4; ++t) {
    const int n = t * 16 + m;
    v16h bv = __builtin_shufflevector(*(const _Float16 __attribute__((ext_vector_type(8)))*)(W1ch + n * 32 + kbase), *(const _Float16 __attribute__((ext_vector_type(8)))*)(W1ch + n * 32 + 16 + kbase), 0,1,2,3,4,5,6,7,8,9,10,11,12,13,14,15);
    v8f c = {};
    c = __builtin_amdgcn_wmma_f32_16x16x32_f16(false, a, false, bv, (short)0, c, false, false);

    const int col = t * 16 + m;          // padded params -> zeros for col>=60
    const float bb  = prm[3 * 64 + col];
    const float s1  = prm[4 * 64 + col];
    const float o1  = prm[5 * 64 + col];
    const float s2  = prm[6 * 64 + col];
    const float o2  = prm[7 * 64 + col];
    const float w40 = prm[8 * 64 + col];
    const float w41 = prm[9 * 64 + col];
    float zv[8];
    for (int v = 0; v < 8; ++v) zv[v] = fmaxf((c[v] + bb) * s1 + o1, 0.0f);
    float mA = fmaxf(fmaxf(zv[0], zv[1]), fmaxf(zv[2], zv[3]));
    float mB = fmaxf(fmaxf(zv[4], zv[5]), fmaxf(zv[6], zv[7]));
    float zA = fmaxf(mA * s2 + o2, 0.0f);
    float zB = fmaxf(mB * s2 + o2, 0.0f);
    accA0 += zA * w40; accA1 += zA * w41;
    accB0 += zB * w40; accB1 += zB * w41;
  }
  for (int d = 1; d < 16; d <<= 1) {
    accA0 += __shfl_xor(accA0, d, 32);
    accA1 += __shfl_xor(accA1, d, 32);
    accB0 += __shfl_xor(accB0, d, 32);
    accB1 += __shfl_xor(accB1, d, 32);
  }
  if (m == 0) {
    const int p = mtile * 4 + hi * 2;  // 4 points per 16-row tile
    out[(size_t)p * 2 + 0]       = accA0 + b4[0];
    out[(size_t)p * 2 + 1]       = accA1 + b4[1];
    out[(size_t)(p + 1) * 2 + 0] = accB0 + b4[0];
    out[(size_t)(p + 1) * 2 + 1] = accB1 + b4[1];
  }
}

extern "C" void kernel_launch(void* const* d_in, const int* in_sizes, int n_in,
                              void* d_out, int out_size, void* d_ws, size_t ws_size,
                              hipStream_t stream) {
  (void)in_sizes; (void)n_in; (void)out_size; (void)ws_size;
  const float* x    = (const float*)d_in[0];
  const float* W1   = (const float*)d_in[1];
  const float* b1   = (const float*)d_in[2];
  const float* g1   = (const float*)d_in[3];
  const float* be1  = (const float*)d_in[4];
  const float* Wc0  = (const float*)d_in[5];
  const float* bc0  = (const float*)d_in[6];
  const float* gc0  = (const float*)d_in[7];
  const float* bec0 = (const float*)d_in[8];
  const float* Wc1  = (const float*)d_in[9];
  const float* bc1  = (const float*)d_in[10];
  const float* gc1  = (const float*)d_in[11];
  const float* bec1 = (const float*)d_in[12];
  const float* g2   = (const float*)d_in[13];
  const float* be2  = (const float*)d_in[14];
  const float* W4   = (const float*)d_in[15];
  const float* b4   = (const float*)d_in[16];

  char* ws = (char*)d_ws;
  float*     h     = (float*)(ws + OFF_H);
  float*     hW    = (float*)(ws + OFF_HW);
  float*     gterm = (float*)(ws + OFF_GTERM);
  int*       idx   = (int*)(ws + OFF_IDX);
  _Float16*  z0    = (_Float16*)(ws + OFF_Z0);
  _Float16*  W1h   = (_Float16*)(ws + OFF_W1H);
  _Float16*  W0h   = (_Float16*)(ws + OFF_W0H);
  _Float16*  W1ch  = (_Float16*)(ws + OFF_W1CH);
  float*     prm   = (float*)(ws + OFF_PRM);
  float*     out   = (float*)d_out;

  k_pack<<<dim3(1), dim3(256), 0, stream>>>(W1, b1, g1, be1, Wc0, Wc1, bc1, gc1,
                                            bec1, g2, be2, W4, W1h, W0h, W1ch, prm);
  k_embed_gemm1<<<dim3(ROWS1 / 64), dim3(128), 0, stream>>>(x, W1h, prm, h);
  k_gmax_gterm<<<dim3(Bsz), dim3(64), 0, stream>>>(h, Wc0, bc0, gterm);
  k_ballquery<<<dim3(Bsz * (Npts / 256)), dim3(256), 0, stream>>>(x, idx);
  k_hW<<<dim3(ROWS1 / 64), dim3(128), 0, stream>>>(h, W0h, hW);
  k_group_z0<<<dim3(ROWS2 / 256), dim3(256), 0, stream>>>(x, idx, hW, gterm, Wc0, gc0, bec0, z0);
  k_z1_out<<<dim3(ROWS2 / 64), dim3(128), 0, stream>>>(z0, W1ch, prm, b4, out);
}
